// MultiheadAttention_29205777612840
// MI455X (gfx1250) — matrix operand order repaired and checked by reading, not compiled, not run on hardware
//
#include <hip/hip_runtime.h>
#include <hip/hip_bf16.h>

// ---------------------------------------------------------------------------
// MI455X / gfx1250 multi-head causal attention
// bf16 WMMA (f32 acc) + TDM tensor_load_to_lds (K/V) + async LDS copies (GEMM)
// S^T = K*Q^T formulation: softmax stats are per-lane, 1 shuffle per reduction
// ---------------------------------------------------------------------------

typedef __bf16 bf16;
typedef __attribute__((ext_vector_type(16))) __bf16 v16bf;
typedef __attribute__((ext_vector_type(8)))  __bf16 bf16x8;  // 16 bytes
typedef __attribute__((ext_vector_type(4)))  __bf16 bf16x4;  // 8 bytes
typedef __attribute__((ext_vector_type(8)))  float  v8f;
typedef __attribute__((ext_vector_type(4)))  float  f32x4;
typedef __attribute__((ext_vector_type(4)))  unsigned int u32x4;
typedef __attribute__((ext_vector_type(8)))  int    i32x8;
typedef __attribute__((ext_vector_type(4)))  int    i32x4;

#define WMMA_BF16(a, b, c) \
  __builtin_amdgcn_wmma_f32_16x16x32_bf16(false, (a), false, (b), (short)0, (c), false, false)

static constexpr int N_SEQ  = 2048;
static constexpr int DM     = 1024;   // d_model = H*D
static constexpr int NH     = 16;
static constexpr int DH     = 64;
static constexpr int BATCH  = 2;
static constexpr int M_TOT  = BATCH * N_SEQ;  // 4096 rows for all projections

// --- LDS addressing + CDNA5 async copy (ASYNCcnt) --------------------------
__device__ inline unsigned lds_off(const void* p) {
  // flat shared pointer -> LDS byte offset (aperture: LDS address is the low 32 bits)
  return (unsigned)(unsigned long long)p;
}
__device__ inline void async_copy_b128(unsigned dst_lds, const void* src_global) {
  asm volatile("global_load_async_to_lds_b128 %0, %1, off"
               :: "v"(dst_lds), "v"(src_global) : "memory");
}
__device__ inline void wait_asynccnt0() {
  asm volatile("s_wait_asynccnt 0" ::: "memory");
}

// --- CDNA5 Tensor Data Mover: 2D 32x64 bf16 tile, global -> LDS (TENSORcnt)
// D# per ISA 8.3-8.5: count=1, type=2(image), data_size=1(2B),
// tensor dim0=64 / dim0_stride=1024, tile 64x32,
// pad_enable: +4 dwords after every 32 dwords -> 144B LDS row pitch.
// This toolchain exposes the 6-arg builtin: (g0, g1, g2, g3, g4, cpol).
__device__ inline void tdm_load_tile_32x64(unsigned lds_addr, const void* gptr) {
  unsigned long long ga = (unsigned long long)gptr;
  u32x4 g0;
  g0[0] = 1u;                                            // count=1 (valid), user
  g0[1] = lds_addr;                                      // LDS byte address
  g0[2] = (unsigned)(ga & 0xFFFFFFFFu);                  // global_addr[31:0]
  g0[3] = (unsigned)((ga >> 32) & 0x01FFFFFFu) | (2u << 30);  // [56:32] | type=2
  i32x8 g1;
  g1[0] = (int)((1u << 16) | (1u << 20) | (4u << 22) | (3u << 25));
  //            data_size=2B | pad_en | pad_interval=32dw | pad_amount=4dw
  g1[1] = (int)(64u << 16);          // tensor_dim0 = 64        (bits 79:48 lo)
  g1[2] = (int)(32u << 16);          // dim0 hi=0 | tensor_dim1 = 32
  g1[3] = (int)(64u << 16);          // dim1 hi=0 | tile_dim0 = 64
  g1[4] = (int)32u;                  // tile_dim1 = 32, tile_dim2 = 0
  g1[5] = (int)1024u;                // tensor_dim0_stride = DM elems (lo)
  g1[6] = 0;                         // stride hi | dim1_stride lo (unused, 2D)
  g1[7] = 0;
  i32x4 z4 = {0, 0, 0, 0};           // groups 2/3 unused (2D tensor)
  i32x8 z8 = {0, 0, 0, 0, 0, 0, 0, 0};
  __builtin_amdgcn_tensor_load_to_lds(g0, g1, z4, z4, z8, 0);
}

// --- fragment loaders matching CDNA5 WMMA VGPR layouts (ISA 7.12.2) --------
// A (16x32 bf16): lane holds row M=lane%16; elems 0-7 -> K = kbase+8*half..,
//                 elems 8-15 -> K = kbase+16+8*half..
__device__ inline v16bf load_a16(const bf16* rowptr, int half, int kbase) {
  bf16x8 lo = *(const bf16x8*)(rowptr + kbase + 8 * half);
  bf16x8 hi = *(const bf16x8*)(rowptr + kbase + 16 + 8 * half);
  v16bf a;
#pragma unroll
  for (int i = 0; i < 8; ++i) { a[i] = lo[i]; a[i + 8] = hi[i]; }
  return a;
}
// B (32x16 bf16): lane holds col N=lane%16; elems 0-7 -> K = 8*half..,
//                 elems 8-15 -> K = 16+8*half.. (rows strided by pitch)
__device__ inline v16bf load_b16(const bf16* p, int pitch, int half) {
  v16bf b;
#pragma unroll
  for (int i = 0; i < 8; ++i) { b[i] = p[(8 * half + i) * pitch]; b[i + 8] = p[(16 + 8 * half + i) * pitch]; }
  return b;
}

// --- f32 -> bf16 conversion pass -------------------------------------------
__global__ void cvt_f32_bf16(const float* __restrict__ in, bf16* __restrict__ out, int n4) {
  int i = blockIdx.x * blockDim.x + threadIdx.x;
  if (i < n4) {
    f32x4 v = *(const f32x4*)(in + (size_t)i * 4);
    bf16x4 o;
#pragma unroll
    for (int j = 0; j < 4; ++j) o[j] = (bf16)v[j];
    *(bf16x4*)(out + (size_t)i * 4) = o;
  }
}

// --- GEMM: C[M,1024] = (A[M,1024] @ B[1024,1024] + bias) * scale -----------
// block = 256 threads = 8 waves; tile 128(M) x 64(N).
// W tile double-buffered in LDS via async global->LDS copies (ASYNCcnt).
template <typename OutT>
__global__ __launch_bounds__(256, 2)
void gemm_bias_kernel(const bf16* __restrict__ A, const bf16* __restrict__ B,
                      const float* __restrict__ bias, OutT* __restrict__ C,
                      float scale) {
  __shared__ __align__(16) bf16 ldsB[2][32][72];
  const int tid  = threadIdx.x;
  const int wave = tid >> 5, lane = tid & 31;
  const int half = lane >> 4, l16 = lane & 15;
  const int mBase = blockIdx.x * 128 + wave * 16;
  const int nBase = blockIdx.y * 64;
  const bf16* arow = A + (size_t)(mBase + l16) * DM;

  v8f acc[4];
#pragma unroll
  for (int t = 0; t < 4; ++t)
#pragma unroll
    for (int i = 0; i < 8; ++i) acc[t][i] = 0.0f;

  const int kk = tid >> 3;        // 0..31 (K row of the W tile)
  const int nn = (tid & 7) * 8;   // 0..56 (N offset, 16B chunk)
  const bf16* gsrc = B + (size_t)kk * DM + nBase + nn;

  async_copy_b128(lds_off(&ldsB[0][kk][nn]), gsrc);   // prologue: tile 0
  int buf = 0;

  for (int k0 = 0; k0 < DM; k0 += 32) {
    wait_asynccnt0();   // this wave's copy of the current tile has landed
    __syncthreads();    // ... and everyone else's
    if (k0 + 32 < DM)   // overlap next tile's copy with this tile's WMMAs
      async_copy_b128(lds_off(&ldsB[buf ^ 1][kk][nn]), gsrc + (size_t)(k0 + 32) * DM);

    v16bf afrag = load_a16(arow, half, k0);
#pragma unroll
    for (int t = 0; t < 4; ++t) {
      v16bf bfrag = load_b16(&ldsB[buf][0][t * 16 + l16], 72, half);
      acc[t] = WMMA_BF16(afrag, bfrag, acc[t]);
    }
    __syncthreads();    // all waves done reading buf before it is overwritten
    buf ^= 1;
  }

  // epilogue: C layout lane -> (M = r + 8*half, N = l16)
#pragma unroll
  for (int t = 0; t < 4; ++t) {
    int col = nBase + t * 16 + l16;
    float bv = bias[col];
#pragma unroll
    for (int r = 0; r < 8; ++r) {
      int row = mBase + r + 8 * half;
      float v = (acc[t][r] + bv) * scale;
      C[(size_t)row * DM + col] = static_cast<OutT>(v);
    }
  }
}

// --- causal flash attention, S^T = K Q^T formulation ------------------------
// block = 256 threads = 8 waves; block owns (batch, head, 128 query rows);
// each wave owns 16 query rows. Key tiles of 32, double-buffered, fetched by
// the Tensor Data Mover (wave 0 issues DMA, TENSORcnt + barrier to publish).
// C layout of S^T: M = key, N = query -> softmax stats are per-lane scalars.
// 1/sqrt(D) is pre-folded into Qh.
__global__ __launch_bounds__(256, 2)
void attention_kernel(const bf16* __restrict__ Qh, const bf16* __restrict__ Kh,
                      const bf16* __restrict__ Vh, bf16* __restrict__ Oh) {
  __shared__ __align__(16) bf16 ldsK[2][32][72];    // [key][d], 144B pitch (TDM pad)
  __shared__ __align__(16) bf16 ldsV[2][32][72];    // [key][d], 144B pitch (TDM pad)
  __shared__ __align__(16) bf16 ldsQT[8][64][16];   // per-wave Q^T [d][query]
  __shared__ __align__(16) bf16 ldsP[8][16][32];    // per-wave P [query][key]
  __shared__ __align__(16) float ldsStat[8][16];    // per-wave per-query scalars

  const int tid  = threadIdx.x;
  const int wave = tid >> 5, lane = tid & 31;
  const int half = lane >> 4, l16 = lane & 15;
  const int batch = blockIdx.z, head = blockIdx.y;
  const int blockRow0 = blockIdx.x * 128;
  const int qRow0 = blockRow0 + wave * 16;
  const size_t seqBase = (size_t)batch * N_SEQ;

  // --- one-time Q^T staging: lane covers row l16, d in [32*half, 32*half+32)
  {
    const bf16* qrow = Qh + (seqBase + qRow0 + l16) * DM + (size_t)head * DH + 32 * half;
    bf16x8 q0 = *(const bf16x8*)(qrow);
    bf16x8 q1 = *(const bf16x8*)(qrow + 8);
    bf16x8 q2 = *(const bf16x8*)(qrow + 16);
    bf16x8 q3 = *(const bf16x8*)(qrow + 24);
#pragma unroll
    for (int i = 0; i < 8; ++i) {
      ldsQT[wave][32 * half + i][l16]      = q0[i];
      ldsQT[wave][32 * half + 8 + i][l16]  = q1[i];
      ldsQT[wave][32 * half + 16 + i][l16] = q2[i];
      ldsQT[wave][32 * half + 24 + i][l16] = q3[i];
    }
  }
  asm volatile("s_wait_dscnt 0" ::: "memory");  // wave-local transpose fence
  // persistent Q^T B-fragments (d rows 0..31 and 32..63, 16 query cols)
  v16bf bq0 = load_b16(&ldsQT[wave][0][l16], 16, half);
  v16bf bq1 = load_b16(&ldsQT[wave][32][l16], 16, half);

  float m_i = -1e30f, l_i = 0.0f;   // per-lane: stats of query column l16
  v8f accO[4];
#pragma unroll
  for (int t = 0; t < 4; ++t)
#pragma unroll
    for (int i = 0; i < 8; ++i) accO[t][i] = 0.0f;

  const int jEnd = blockRow0 + 128;  // causal: no keys beyond block's last row
  const bf16* kTile = Kh + seqBase * DM + (size_t)head * DH;
  const bf16* vTile = Vh + seqBase * DM + (size_t)head * DH;

  // prologue: TDM pulls tile 0 into buffer 0
  if (wave == 0) {
    tdm_load_tile_32x64(lds_off(&ldsK[0][0][0]), kTile);
    tdm_load_tile_32x64(lds_off(&ldsV[0][0][0]), vTile);
  }
  int buf = 0;

  for (int j0 = 0; j0 < jEnd; j0 += 32) {
    if (wave == 0) __builtin_amdgcn_s_wait_tensorcnt(0);  // DMA landed
    __syncthreads();                                      // publish to all waves

    const int jn = j0 + 32;
    if (jn < jEnd) {
      if (wave == 0) {  // overlap next tile's DMA with this tile's compute
        tdm_load_tile_32x64(lds_off(&ldsK[buf ^ 1][0][0]), kTile + (size_t)jn * DM);
        tdm_load_tile_32x64(lds_off(&ldsV[buf ^ 1][0][0]), vTile + (size_t)jn * DM);
      }
      if (wave == 1 && jn + 32 < jEnd)  // warm L2 two tiles ahead for the TDM
        __builtin_prefetch(kTile + (size_t)(jn + 32 + lane) * DM, 0, 1);
    }

    if (j0 <= qRow0 + 15) {  // tile intersects this wave's causal region
      // S^T = K Q^T : two 16-key subtiles, K-dim = d = 64 (2 chunks)
      v8f st[2];
#pragma unroll
      for (int kt = 0; kt < 2; ++kt) {
#pragma unroll
        for (int i = 0; i < 8; ++i) st[kt][i] = 0.0f;
        v16bf ak0 = load_a16(&ldsK[buf][kt * 16 + l16][0], half, 0);
        st[kt] = WMMA_BF16(ak0, bq0, st[kt]);
        v16bf ak1 = load_a16(&ldsK[buf][kt * 16 + l16][0], half, 32);
        st[kt] = WMMA_BF16(ak1, bq1, st[kt]);
      }
      // causal mask: element (key = j0+kt*16+r+8*half, query = qRow0+l16)
      const int qAbs = qRow0 + l16;
#pragma unroll
      for (int kt = 0; kt < 2; ++kt) {
#pragma unroll
        for (int r = 0; r < 8; ++r) {
          int key = j0 + kt * 16 + r + 8 * half;
          st[kt][r] = (key > qAbs) ? -1e30f : st[kt][r];
        }
      }
      // online softmax: lane holds 16 of the 32 keys of query l16;
      // the other 16 live in the paired lane (half flipped) -> 1 shuffle.
      float mt = -1e30f;
#pragma unroll
      for (int kt = 0; kt < 2; ++kt)
#pragma unroll
        for (int r = 0; r < 8; ++r) mt = fmaxf(mt, st[kt][r]);
      mt = fmaxf(mt, __shfl_xor(mt, 16, 32));
      float mnew = fmaxf(m_i, mt);
      float corr = __expf(m_i - mnew);
      float rs = 0.0f;
      bf16x8 pk0, pk1;
#pragma unroll
      for (int r = 0; r < 8; ++r) { float pp = __expf(st[0][r] - mnew); rs += pp; pk0[r] = (bf16)pp; }
#pragma unroll
      for (int r = 0; r < 8; ++r) { float pp = __expf(st[1][r] - mnew); rs += pp; pk1[r] = (bf16)pp; }
      // P relayout: keys r+8*half are consecutive -> two aligned b128 stores
      *(bf16x8*)(&ldsP[wave][l16][8 * half])      = pk0;
      *(bf16x8*)(&ldsP[wave][l16][16 + 8 * half]) = pk1;
      rs += __shfl_xor(rs, 16, 32);
      l_i = l_i * corr + rs;
      m_i = mnew;
      if (half == 0) ldsStat[wave][l16] = corr;     // broadcast per-query corr
      asm volatile("s_wait_dscnt 0" ::: "memory");  // wave-local LDS fence

      // rescale O (C layout row = query r+8*half) by corr of that query
      f32x4 c0 = *(const f32x4*)(&ldsStat[wave][8 * half]);
      f32x4 c1 = *(const f32x4*)(&ldsStat[wave][8 * half + 4]);
#pragma unroll
      for (int t = 0; t < 4; ++t)
#pragma unroll
        for (int r = 0; r < 8; ++r) accO[t][r] *= (r < 4) ? c0[r & 3] : c1[r & 3];

      // O += P V : A = P (16q x 32k), B = V tile (32k x 16d per subtile)
      v16bf ap = load_a16(&ldsP[wave][l16][0], half, 0);
#pragma unroll
      for (int t = 0; t < 4; ++t) {
        v16bf bv16 = load_b16(&ldsV[buf][0][t * 16 + l16], 72, half);
        accO[t] = WMMA_BF16(ap, bv16, accO[t]);
      }
    }
    __syncthreads();  // all waves done with buf before TDM overwrites it
    buf ^= 1;
  }

  // normalize by 1/l (per query) and store bf16 attention output [b,n,H*D]
  if (half == 0) ldsStat[wave][l16] = 1.0f / l_i;
  asm volatile("s_wait_dscnt 0" ::: "memory");
  f32x4 i0 = *(const f32x4*)(&ldsStat[wave][8 * half]);
  f32x4 i1 = *(const f32x4*)(&ldsStat[wave][8 * half + 4]);
#pragma unroll
  for (int t = 0; t < 4; ++t) {
#pragma unroll
    for (int r = 0; r < 8; ++r) {
      float o = accO[t][r] * ((r < 4) ? i0[r & 3] : i1[r & 3]);
      Oh[(seqBase + qRow0 + r + 8 * half) * DM + (size_t)head * DH + t * 16 + l16] = (bf16)o;
    }
  }
}

// ---------------------------------------------------------------------------
extern "C" void kernel_launch(void* const* d_in, const int* in_sizes, int n_in,
                              void* d_out, int out_size, void* d_ws, size_t ws_size,
                              hipStream_t stream) {
  (void)in_sizes; (void)n_in; (void)out_size; (void)ws_size;
  const float* q  = (const float*)d_in[0];
  const float* k  = (const float*)d_in[1];
  const float* v  = (const float*)d_in[2];
  const float* Wq = (const float*)d_in[3];
  const float* bq = (const float*)d_in[4];
  const float* Wk = (const float*)d_in[5];
  const float* bk = (const float*)d_in[6];
  const float* Wv = (const float*)d_in[7];
  const float* bv = (const float*)d_in[8];
  const float* Wo = (const float*)d_in[9];
  const float* bo = (const float*)d_in[10];
  float* out = (float*)d_out;

  const size_t nTok = (size_t)M_TOT * DM;  // 4M elems
  const size_t nW   = (size_t)DM * DM;     // 1M elems
  bf16* p = (bf16*)d_ws;
  bf16* qb  = p; p += nTok;
  bf16* kb  = p; p += nTok;
  bf16* vb  = p; p += nTok;
  bf16* Wqb = p; p += nW;
  bf16* Wkb = p; p += nW;
  bf16* Wvb = p; p += nW;
  bf16* Wob = p; p += nW;
  bf16* Qhb = p; p += nTok;
  bf16* Khb = p; p += nTok;
  bf16* Vhb = p; p += nTok;
  bf16* Ab  = p; p += nTok;

  auto cvt = [&](const float* src, bf16* dst, size_t n) {
    int n4 = (int)(n / 4);
    cvt_f32_bf16<<<(n4 + 255) / 256, 256, 0, stream>>>(src, dst, n4);
  };
  cvt(q, qb, nTok);  cvt(k, kb, nTok);  cvt(v, vb, nTok);
  cvt(Wq, Wqb, nW);  cvt(Wk, Wkb, nW);  cvt(Wv, Wvb, nW);  cvt(Wo, Wob, nW);

  dim3 gGemm(M_TOT / 128, DM / 64);
  // 1/sqrt(D)=0.125 folded into Q projection
  gemm_bias_kernel<bf16><<<gGemm, 256, 0, stream>>>(qb, Wqb, bq, Qhb, 0.125f);
  gemm_bias_kernel<bf16><<<gGemm, 256, 0, stream>>>(kb, Wkb, bk, Khb, 1.0f);
  gemm_bias_kernel<bf16><<<gGemm, 256, 0, stream>>>(vb, Wvb, bv, Vhb, 1.0f);

  dim3 gAtt(N_SEQ / 128, NH, BATCH);
  attention_kernel<<<gAtt, 256, 0, stream>>>(Qhb, Khb, Vhb, Ab);

  gemm_bias_kernel<float><<<gGemm, 256, 0, stream>>>(Ab, Wob, bo, out, 1.0f);
}
